// se_BiLSTM_72799695667347
// MI455X (gfx1250) — matrix operand order repaired and checked by reading, not compiled, not run on hardware
//
#include <hip/hip_runtime.h>
#include <hip/hip_bf16.h>

// ---------------------------------------------------------------------------
// BiLSTM for MI455X (gfx1250, wave32, WMMA).
// Phase 1: xz = x@W + b  (bf16 WMMA GEMM, whole chip)
// Phase 2: persistent 1-WG-per-direction recurrent scan, h in LDS, c in VGPRs
// Phase 3: out = h_fwd + h_bwd
// ---------------------------------------------------------------------------

typedef __bf16 bf16;
typedef __attribute__((ext_vector_type(16))) __bf16 v16bf;
typedef __attribute__((ext_vector_type(8)))  __bf16 v8bf;
typedef __attribute__((ext_vector_type(4)))  __bf16 v4bf;
typedef __attribute__((ext_vector_type(8)))  float  v8f;

#define NB   64      // batch
#define NT   1024    // time steps
#define NF   512     // input features
#define NH   256     // hidden units
#define NG   1024    // 4*H gate width
#define BT   65536   // NB*NT rows of the input GEMM

static __device__ __forceinline__ v16bf cat8(v8bf lo, v8bf hi) {
    return __builtin_shufflevector(lo, hi, 0,1,2,3,4,5,6,7,8,9,10,11,12,13,14,15);
}

static __device__ __forceinline__ float sigm(float x) {
    return 1.0f / (1.0f + __expf(-x));
}
static __device__ __forceinline__ float tanh_fast(float x) {
    float e = __expf(-2.0f * fabsf(x));
    float r = (1.0f - e) / (1.0f + e);
    return copysignf(r, x);
}

// ---------------------------------------------------------------------------
// Phase 0: U [H=256][G=1024] f32  ->  Ut [dir][col=1024][k=256] bf16
// (column-major-in-K so B fragments are two contiguous b128 loads per lane)
// ---------------------------------------------------------------------------
__global__ __launch_bounds__(256) void u_transpose(const float* __restrict__ Uf,
                                                   const float* __restrict__ Ub,
                                                   bf16* __restrict__ Ut) {
    int id  = blockIdx.x * 256 + threadIdx.x;   // 0 .. 2*262144-1
    int d   = id >> 18;
    int r   = id & 262143;
    int col = r >> 8;
    int k   = r & 255;
    const float* U = d ? Ub : Uf;
    Ut[id] = (bf16)U[k * NG + col];
}

// ---------------------------------------------------------------------------
// Phase 1: xz[BT][NG] (bf16) = X[BT][NF] @ W[NF][NG] + bias
// 128x128 block tile, K staged in LDS as bf16 (A row-major, B transposed),
// 8 waves, each wave: 64x32 (4x2 tiles of 16x16), K chunks of 32.
// ---------------------------------------------------------------------------
__global__ __launch_bounds__(256) void xw_gemm(const float* __restrict__ X,
                                               const float* __restrict__ W,
                                               const float* __restrict__ bias,
                                               bf16* __restrict__ out) {
    __shared__ bf16 la[128 * 48];   // [row 0..127][k 0..31], pitch 48 (16B aligned)
    __shared__ bf16 lb[128 * 48];   // [n   0..127][k 0..31], pitch 48

    const int tid  = threadIdx.x;
    const int lane = tid & 31;
    const int wave = tid >> 5;
    const int lrow = lane & 15;
    const int lhi  = lane >> 4;
    const int rowBase = blockIdx.x * 128;
    const int nBase   = blockIdx.y * 128;
    const int wm = wave & 1;        // 0..1 -> 64-row half
    const int wn = wave >> 1;       // 0..3 -> 32-col quarter

    v8f acc[4][2];
    for (int s = 0; s < 4; ++s)
        for (int u = 0; u < 2; ++u)
            for (int v = 0; v < 8; ++v) acc[s][u][v] = 0.0f;

    for (int kc = 0; kc < 16; ++kc) {
        const int k0 = kc * 32;
        // stage A: 128x32 f32 -> bf16, row-major
        for (int it = 0; it < 4; ++it) {
            int fi = it * 256 + tid;            // float4 index, 1024 total
            int r  = fi >> 3;
            int cc = (fi & 7) * 4;
            const float4 v4 = *(const float4*)&X[(size_t)(rowBase + r) * NF + k0 + cc];
            v4bf o;
            o[0] = (bf16)v4.x; o[1] = (bf16)v4.y; o[2] = (bf16)v4.z; o[3] = (bf16)v4.w;
            *(v4bf*)&la[r * 48 + cc] = o;
        }
        // stage B: 32x128 f32 -> bf16 transposed to [n][k]
        for (int it = 0; it < 16; ++it) {
            int ei = it * 256 + tid;            // 4096 elements
            int k  = ei >> 7;
            int n  = ei & 127;
            lb[n * 48 + k] = (bf16)W[(size_t)(k0 + k) * NG + nBase + n];
        }
        __syncthreads();

        v16bf af[4], bfr[2];
        for (int s = 0; s < 4; ++s) {
            int base = (wm * 64 + s * 16 + lrow) * 48 + lhi * 8;
            v8bf lo = *(const v8bf*)&la[base];
            v8bf hi = *(const v8bf*)&la[base + 16];
            af[s] = cat8(lo, hi);
        }
        for (int u = 0; u < 2; ++u) {
            int base = (wn * 32 + u * 16 + lrow) * 48 + lhi * 8;
            v8bf lo = *(const v8bf*)&lb[base];
            v8bf hi = *(const v8bf*)&lb[base + 16];
            bfr[u] = cat8(lo, hi);
        }
        for (int s = 0; s < 4; ++s)
            for (int u = 0; u < 2; ++u)
                acc[s][u] = __builtin_amdgcn_wmma_f32_16x16x32_bf16(
                    false, af[s], false, bfr[u], (short)0, acc[s][u], false, false);
        __syncthreads();
    }

    // epilogue: + bias, convert to bf16, store
    for (int u = 0; u < 2; ++u) {
        const int gcol = nBase + wn * 32 + u * 16 + lrow;
        const float bv = bias[gcol];
        for (int s = 0; s < 4; ++s)
            for (int v = 0; v < 8; ++v) {
                const int grow = rowBase + wm * 64 + s * 16 + v + 8 * lhi;
                out[(size_t)grow * NG + gcol] = (bf16)(acc[s][u][v] + bv);
            }
    }
}

// ---------------------------------------------------------------------------
// Phase 2: recurrent scan. grid = 2 blocks (dir), 1024 threads (32 waves).
// Wave w owns column tile j = w>>1 (16 cols of H) and two batch tiles
// m = (w&1)*2 + {0,1}. Each step: z = xz_t + h@U, gates, c/h update.
// h lives in LDS (bf16, padded pitch); c lives in VGPRs for the whole scan.
// ---------------------------------------------------------------------------
#define HPITCH 264    // 256 + 8 pad: 528B row -> 16B aligned, conflict-free

__global__ __launch_bounds__(1024) void lstm_scan(const bf16* __restrict__ xzf,
                                                  const bf16* __restrict__ xzb,
                                                  const bf16* __restrict__ UtAll,
                                                  float* __restrict__ hfin) {
    const int dir = blockIdx.x;
    const bf16* __restrict__ xz = dir ? xzb : xzf;
    const bf16* __restrict__ Ut = UtAll + (size_t)dir * (NG * NH);
    float* __restrict__ hout = hfin + dir * (NB * NH);

    __shared__ bf16 hs[NB * HPITCH];
    for (int i = threadIdx.x; i < NB * HPITCH; i += 1024) hs[i] = (bf16)0.0f;

    const int lane = threadIdx.x & 31;
    const int wave = threadIdx.x >> 5;
    const int lrow = lane & 15;
    const int lhi  = lane >> 4;
    const int j    = wave >> 1;          // 0..15 column tile of H
    const int mp   = (wave & 1) * 2;     // first of two batch tiles

    float c[2][8], hreg[2][8];
    for (int mi = 0; mi < 2; ++mi)
        for (int v = 0; v < 8; ++v) { c[mi][v] = 0.0f; hreg[mi][v] = 0.0f; }

    __syncthreads();

    for (int t = 0; t < NT; ++t) {
        const int tt = dir ? (NT - 1 - t) : t;

        // z := xz_t  (per-element bf16 loads; C-layout row = v+8*lhi, col = lrow)
        v8f z[2][4];
        for (int mi = 0; mi < 2; ++mi) {
            const int m = mp + mi;
            for (int v = 0; v < 8; ++v) {
                const int b = m * 16 + lhi * 8 + v;
                const bf16* p = xz + ((size_t)b * NT + tt) * NG + j * 16 + lrow;
                for (int g = 0; g < 4; ++g)
                    z[mi][g][v] = (float)p[g * 256];
            }
        }

        // z += h @ U   (K = 256 in 8 chunks of 32)
        for (int kc = 0; kc < 8; ++kc) {
            v16bf a[2];
            for (int mi = 0; mi < 2; ++mi) {
                const int base = ((mp + mi) * 16 + lrow) * HPITCH + kc * 32 + lhi * 8;
                v8bf lo = *(const v8bf*)&hs[base];
                v8bf hi = *(const v8bf*)&hs[base + 16];
                a[mi] = cat8(lo, hi);
            }
            for (int g = 0; g < 4; ++g) {
                const bf16* up = Ut + ((size_t)(g * 256 + j * 16 + lrow)) * NH
                                    + kc * 32 + lhi * 8;
                v8bf lo = *(const v8bf*)up;
                v8bf hi = *(const v8bf*)(up + 16);
                v16bf bfrag = cat8(lo, hi);
                for (int mi = 0; mi < 2; ++mi)
                    z[mi][g] = __builtin_amdgcn_wmma_f32_16x16x32_bf16(
                        false, a[mi], false, bfrag, (short)0, z[mi][g], false, false);
            }
        }

        __syncthreads();   // all waves finished reading h of this step

        // gates + state update + write new h (bf16) into LDS
        for (int mi = 0; mi < 2; ++mi) {
            for (int v = 0; v < 8; ++v) {
                const float ig = sigm(z[mi][0][v]);
                const float fg = sigm(z[mi][1][v]);
                const float gg = tanh_fast(z[mi][2][v]);
                const float og = sigm(z[mi][3][v]);
                const float cn = fg * c[mi][v] + ig * gg;
                c[mi][v] = cn;
                const float hv = og * tanh_fast(cn);
                hreg[mi][v] = hv;
                const int row = (mp + mi) * 16 + v + 8 * lhi;
                hs[row * HPITCH + j * 16 + lrow] = (bf16)hv;
            }
        }

        __syncthreads();   // new h visible before next step's reads
    }

    // final hidden state (f32) to workspace
    for (int mi = 0; mi < 2; ++mi)
        for (int v = 0; v < 8; ++v) {
            const int row = (mp + mi) * 16 + v + 8 * lhi;
            hout[row * NH + j * 16 + lrow] = hreg[mi][v];
        }
}

// ---------------------------------------------------------------------------
// Phase 3: out = h_fwd + h_bwd
// ---------------------------------------------------------------------------
__global__ __launch_bounds__(256) void final_sum(const float* __restrict__ hfin,
                                                 float* __restrict__ out) {
    const int i = blockIdx.x * 256 + threadIdx.x;   // 0..16383
    out[i] = hfin[i] + hfin[i + NB * NH];
}

// ---------------------------------------------------------------------------
extern "C" void kernel_launch(void* const* d_in, const int* in_sizes, int n_in,
                              void* d_out, int out_size, void* d_ws, size_t ws_size,
                              hipStream_t stream) {
    const float* x  = (const float*)d_in[0];
    const float* Wf = (const float*)d_in[1];
    const float* Uf = (const float*)d_in[2];
    const float* bf = (const float*)d_in[3];
    const float* Wb = (const float*)d_in[4];
    const float* Ub = (const float*)d_in[5];
    const float* bb = (const float*)d_in[6];
    float* out = (float*)d_out;

    char* ws = (char*)d_ws;
    // [0, 1MB)        : Ut  (2 dirs x 1024 x 256 bf16)
    // [1MB, 1MB+128K) : hfin (2 dirs x 64 x 256 f32)
    // then            : xzf, xzb (each 64*1024*1024 bf16 = 128 MiB)
    bf16*  Ut   = (bf16*)ws;
    float* hfin = (float*)(ws + (size_t)2 * NG * NH * sizeof(bf16));
    bf16*  xzf  = (bf16*)(ws + (size_t)2 * NG * NH * sizeof(bf16)
                             + (size_t)2 * NB * NH * sizeof(float));
    bf16*  xzb  = xzf + (size_t)BT * NG;

    u_transpose<<<2048, 256, 0, stream>>>(Uf, Ub, Ut);

    dim3 gg(BT / 128, NG / 128);
    xw_gemm<<<gg, 256, 0, stream>>>(x, Wf, bf, xzf);
    xw_gemm<<<gg, 256, 0, stream>>>(x, Wb, bb, xzb);

    lstm_scan<<<2, 1024, 0, stream>>>(xzf, xzb, Ut, hfin);

    final_sum<<<NB * NH / 256, 256, 0, stream>>>(hfin, out);
}
